// HeteroGGNN_47794396070376
// MI455X (gfx1250) — matrix operand order repaired and checked by reading, not compiled, not run on hardware
//
#include <hip/hip_runtime.h>
#include <stdint.h>

// ---------------------------------------------------------------------------
// HeteroGGNN (GGNN item->item + GAT item<->session), L=2, H=128
// GEMMs in bf16 via v_wmma_f32_16x16x32_bf16; edge ops via fp32 atomics.
// ---------------------------------------------------------------------------

enum { NI = 100000, NS = 20000, H = 128, H3 = 384, NE = 800000, NL = 2, OS = 384 };

typedef __bf16 bf16_t;
typedef bf16_t v16bf __attribute__((ext_vector_type(16)));
typedef bf16_t v8bf  __attribute__((ext_vector_type(8)));
typedef float  v8f   __attribute__((ext_vector_type(8)));

// ---------------- helpers ----------------

__device__ __forceinline__ uint16_t f32_to_bf16_bits(float f) {
  uint32_t u = __float_as_uint(f);
  u += 0x7FFFu + ((u >> 16) & 1u);   // round-to-nearest-even
  return (uint16_t)(u >> 16);
}

// monotonic float<->uint mapping for atomicMax-based segment max
__device__ __forceinline__ uint32_t enc_f32(float f) {
  uint32_t u = __float_as_uint(f);
  return (u & 0x80000000u) ? ~u : (u | 0x80000000u);
}
__device__ __forceinline__ float dec_f32(uint32_t e) {
  uint32_t u = (e & 0x80000000u) ? (e & 0x7FFFFFFFu) : ~e;
  return __uint_as_float(u);
}

// ---------------- conversion kernels ----------------

__global__ void k_f32_to_bf16(const float* __restrict__ in,
                              uint16_t* __restrict__ out, int n) {
  int i = blockIdx.x * blockDim.x + threadIdx.x;
  if (i < n) out[i] = f32_to_bf16_bits(in[i]);
}

// W [K x N] f32 -> Wt [N x K] bf16  (K = N = 128 here)
__global__ void k_transpose_bf16(const float* __restrict__ W,
                                 uint16_t* __restrict__ Wt, int K, int N) {
  int i = blockIdx.x * blockDim.x + threadIdx.x;
  if (i >= K * N) return;
  int n = i / K, k = i - n * K;
  Wt[n * K + k] = f32_to_bf16_bits(W[k * N + n]);
}

// ---------------- WMMA GEMM: C[MxN] = A_bf16[MxK=128] * (Bt_bf16[NxK])^T ----
// One wave computes a 16(M) x 64(N) strip with 4 accumulators.
// A fragment layout (16x32 bf16):  lane<16: K = ks+{0..7,16..23};
//                                  lane>=16: K = ks+{8..15,24..31}; row = lane&15
// B fragment layout (32x16 bf16):  col = lane&15; K same as A fragment
// C layout (16x16 f32):            row = v + (lane>=16?8:0); col = lane&15

__global__ void k_gemm_bf16_wmma(const bf16_t* __restrict__ A,
                                 const bf16_t* __restrict__ Bt,
                                 float* __restrict__ C, int M, int N) {
  const int K = 128;
  int lane = threadIdx.x & 31;
  int wave = blockIdx.x * (blockDim.x >> 5) + (threadIdx.x >> 5);
  int ntiles = N >> 6;
  int totalWaves = (M >> 4) * ntiles;
  if (wave >= totalWaves) return;
  int mt = wave / ntiles;
  int nt = wave - mt * ntiles;
  int nbase = nt << 6;
  int halfsel = lane >> 4;             // 0 or 1
  int row = (mt << 4) + (lane & 15);

  v8f acc[4] = {v8f{}, v8f{}, v8f{}, v8f{}};
  const bf16_t* arow = A + (size_t)row * K;

  for (int ks = 0; ks < 128; ks += 32) {
    int ka = ks + (halfsel ? 8 : 0);
    v8bf alo = *(const v8bf*)(arow + ka);
    v8bf ahi = *(const v8bf*)(arow + ka + 16);
    v16bf afrag;
#pragma unroll
    for (int i = 0; i < 8; ++i) { afrag[i] = alo[i]; afrag[i + 8] = ahi[i]; }

    int kb = ks + (halfsel ? 8 : 0);
#pragma unroll
    for (int j = 0; j < 4; ++j) {
      const bf16_t* bcol = Bt + (size_t)(nbase + (j << 4) + (lane & 15)) * K + kb;
      v8bf blo = *(const v8bf*)(bcol);
      v8bf bhi = *(const v8bf*)(bcol + 16);
      v16bf bfrag;
#pragma unroll
      for (int i = 0; i < 8; ++i) { bfrag[i] = blo[i]; bfrag[i + 8] = bhi[i]; }
      acc[j] = __builtin_amdgcn_wmma_f32_16x16x32_bf16(
          false, afrag, false, bfrag, (short)0, acc[j], false, false);
    }
  }

  int m0 = (mt << 4) + (halfsel ? 8 : 0);
  int ncol = lane & 15;
#pragma unroll
  for (int v = 0; v < 8; ++v) {
    float* crow = C + (size_t)(m0 + v) * N + nbase + ncol;
    crow[0]  = acc[0][v];
    crow[16] = acc[1][v];
    crow[32] = acc[2][v];
    crow[48] = acc[3][v];
  }
}

// ---------------- attention score: s[i] = dot(X[i,:], a) -------------------

__global__ void k_rowdot(const float* __restrict__ X, const float* __restrict__ a,
                         float* __restrict__ s, int M) {
  int t = blockIdx.x * blockDim.x + threadIdx.x;
  int row = t >> 5, lane = t & 31;
  if (row >= M) return;
  float acc = 0.f;
#pragma unroll
  for (int k = 0; k < 4; ++k) {
    int j = lane + (k << 5);
    acc += X[(size_t)row * H + j] * a[j];
  }
#pragma unroll
  for (int off = 16; off > 0; off >>= 1) acc += __shfl_xor(acc, off, 32);
  if (lane == 0) s[row] = acc;
}

// ---------------- edge kernels ---------------------------------------------

__global__ void k_init_max_den(uint32_t* __restrict__ mmax,
                               float* __restrict__ den, int n) {
  int i = blockIdx.x * blockDim.x + threadIdx.x;
  if (i < n) { mmax[i] = 0x007FFFFFu; /* enc(-inf) */ den[i] = 0.f; }
}

__global__ void k_edge_logit_max(const float* __restrict__ ssrc,
                                 const float* __restrict__ sdst,
                                 const int* __restrict__ src,
                                 const int* __restrict__ dst,
                                 float* __restrict__ ebuf,
                                 uint32_t* __restrict__ mmax, int E) {
  int i = blockIdx.x * blockDim.x + threadIdx.x;
  if (i >= E) return;
  float v = ssrc[src[i]] + sdst[dst[i]];
  float e = (v < 0.f) ? 0.2f * v : v;   // leaky_relu 0.2
  ebuf[i] = e;
  atomicMax(&mmax[dst[i]], enc_f32(e));
}

__global__ void k_edge_exp_den(float* __restrict__ ebuf,
                               const uint32_t* __restrict__ mmax,
                               const int* __restrict__ dst,
                               float* __restrict__ den, int E) {
  int i = blockIdx.x * blockDim.x + threadIdx.x;
  if (i >= E) return;
  int d = dst[i];
  float ex = __expf(ebuf[i] - dec_f32(mmax[d]));
  ebuf[i] = ex;
  atomicAdd(&den[d], ex);
}

// acc[dst,:] += proj[src,:] * (ex / (den[dst]+1e-16)) ; one wave per edge
__global__ void k_gat_scatter(const float* __restrict__ proj,
                              const float* __restrict__ ebuf,
                              const float* __restrict__ den,
                              const int* __restrict__ src,
                              const int* __restrict__ dst,
                              float* __restrict__ acc, int E) {
  int t = blockIdx.x * blockDim.x + threadIdx.x;
  int e = t >> 5, lane = t & 31;
  if (e >= E) return;
  int s = src[e], d = dst[e];
  float alpha = ebuf[e] / (den[d] + 1e-16f);
  const float* prow = proj + (size_t)s * H;
  float* arow = acc + (size_t)d * H;
#pragma unroll
  for (int k = 0; k < 4; ++k) {
    int j = lane + (k << 5);
    atomicAdd(&arow[j], prow[j] * alpha);
  }
}

// m[dst,:] += xw[src,:] ; one wave per edge
__global__ void k_seg_scatter(const float* __restrict__ xw,
                              const int* __restrict__ src,
                              const int* __restrict__ dst,
                              float* __restrict__ acc, int E) {
  int t = blockIdx.x * blockDim.x + threadIdx.x;
  int e = t >> 5, lane = t & 31;
  if (e >= E) return;
  const float* prow = xw + (size_t)src[e] * H;
  float* arow = acc + (size_t)dst[e] * H;
#pragma unroll
  for (int k = 0; k < 4; ++k) {
    int j = lane + (k << 5);
    atomicAdd(&arow[j], prow[j]);
  }
}

// ---------------- GRU cell (GatedGraphConv update) -------------------------

__global__ void k_gru(const float* __restrict__ gi, const float* __restrict__ gh,
                      const float* __restrict__ bih, const float* __restrict__ bhh,
                      const float* __restrict__ x, float* __restrict__ out, int n) {
  int idx = blockIdx.x * blockDim.x + threadIdx.x;
  if (idx >= n) return;
  int i = idx >> 7, j = idx & 127;
  size_t b = (size_t)i * H3;
  float ir = gi[b + j]         + bih[j],         hr = gh[b + j]         + bhh[j];
  float iz = gi[b + H + j]     + bih[H + j],     hz = gh[b + H + j]     + bhh[H + j];
  float ig = gi[b + 2 * H + j] + bih[2 * H + j], hg = gh[b + 2 * H + j] + bhh[2 * H + j];
  float r = 1.f / (1.f + __expf(-(ir + hr)));
  float z = 1.f / (1.f + __expf(-(iz + hz)));
  float g = tanhf(ig + r * hg);
  out[idx] = (1.f - z) * g + z * x[idx];
}

// ---------------- combine / copy-out ---------------------------------------
// v = (a + b? + bias?) ; optional relu ; write next-layer buffer + d_out slot
__global__ void k_combine_store(const float* __restrict__ a,
                                const float* __restrict__ b,
                                const float* __restrict__ bias,
                                float* __restrict__ cur,
                                float* __restrict__ dout,
                                int rows, int colOff, int doRelu) {
  int idx = blockIdx.x * blockDim.x + threadIdx.x;
  if (idx >= rows * H) return;
  int i = idx >> 7, j = idx & 127;
  float v = a[idx];
  if (b) v += b[idx];
  if (bias) v += bias[j];
  if (doRelu) v = fmaxf(v, 0.f);
  cur[idx] = v;
  dout[(size_t)i * OS + colOff + j] = v;
}

// ---------------------------------------------------------------------------

extern "C" void kernel_launch(void* const* d_in, const int* in_sizes, int n_in,
                              void* d_out, int out_size, void* d_ws, size_t ws_size,
                              hipStream_t stream) {
  (void)in_sizes; (void)n_in; (void)out_size; (void)ws_size;

  const float* x_item  = (const float*)d_in[0];
  const float* x_sess  = (const float*)d_in[1];
  const int* ii_src = (const int*)d_in[2];
  const int* ii_dst = (const int*)d_in[3];
  const int* is_src = (const int*)d_in[4];
  const int* is_dst = (const int*)d_in[5];
  const int* si_src = (const int*)d_in[6];
  const int* si_dst = (const int*)d_in[7];
  const float* gg_W    = (const float*)d_in[8];
  const float* gg_Wih  = (const float*)d_in[9];
  const float* gg_Whh  = (const float*)d_in[10];
  const float* gg_bih  = (const float*)d_in[11];
  const float* gg_bhh  = (const float*)d_in[12];
  const float* gis_Wsrc = (const float*)d_in[13];
  const float* gis_Wdst = (const float*)d_in[14];
  const float* gis_asrc = (const float*)d_in[15];
  const float* gis_adst = (const float*)d_in[16];
  const float* gis_b    = (const float*)d_in[17];
  const float* gsi_Wsrc = (const float*)d_in[18];
  const float* gsi_Wdst = (const float*)d_in[19];
  const float* gsi_asrc = (const float*)d_in[20];
  const float* gsi_adst = (const float*)d_in[21];
  const float* gsi_b    = (const float*)d_in[22];

  float* out_items = (float*)d_out;
  float* out_sess  = out_items + (size_t)NI * OS;

  // ---- workspace carve (~770 MB) ----
  char* wp = (char*)d_ws;
  auto carve = [&](size_t bytes) -> void* {
    void* r = (void*)wp;
    wp += (bytes + 255) & ~(size_t)255;
    return r;
  };
  float* hiA  = (float*)carve((size_t)NI * H * 4);
  float* hiB  = (float*)carve((size_t)NI * H * 4);
  float* hsA  = (float*)carve((size_t)NS * H * 4);
  float* hsB  = (float*)carve((size_t)NS * H * 4);
  float* xw   = (float*)carve((size_t)NI * H * 4);   // xW, then GGNN output
  float* mseg = (float*)carve((size_t)NI * H * 4);
  float* gi   = (float*)carve((size_t)NI * H3 * 4);
  float* gh   = (float*)carve((size_t)NI * H3 * 4);
  float* projS = (float*)carve((size_t)NI * H * 4);
  float* projD = (float*)carve((size_t)NI * H * 4);
  float* accI  = (float*)carve((size_t)NI * H * 4);
  float* accS  = (float*)carve((size_t)NS * H * 4);
  float* ssrc  = (float*)carve((size_t)NI * 4);
  float* sdst  = (float*)carve((size_t)NI * 4);
  uint32_t* mmax = (uint32_t*)carve((size_t)NI * 4);
  float* den   = (float*)carve((size_t)NI * 4);
  float* ebuf  = (float*)carve((size_t)NE * 4);
  uint16_t* hi_bf = (uint16_t*)carve((size_t)NI * H * 2);
  uint16_t* hs_bf = (uint16_t*)carve((size_t)NS * H * 2);
  uint16_t* m_bf  = (uint16_t*)carve((size_t)NI * H * 2);
  uint16_t* w_ggT   = (uint16_t*)carve((size_t)NL * H * H * 2);
  uint16_t* w_ih    = (uint16_t*)carve((size_t)NL * H3 * H * 2);
  uint16_t* w_hh    = (uint16_t*)carve((size_t)NL * H3 * H * 2);
  uint16_t* w_gis_s = (uint16_t*)carve((size_t)NL * H * H * 2);
  uint16_t* w_gis_d = (uint16_t*)carve((size_t)NL * H * H * 2);
  uint16_t* w_gsi_s = (uint16_t*)carve((size_t)NL * H * H * 2);
  uint16_t* w_gsi_d = (uint16_t*)carve((size_t)NL * H * H * 2);

  const int T = 256;
  auto blk = [](long long n, int t) { return (unsigned)((n + t - 1) / t); };

  auto gemm = [&](const uint16_t* A, const uint16_t* Bt, float* C, int M, int N) {
    long long waves = (long long)(M >> 4) * (N >> 6);
    k_gemm_bf16_wmma<<<blk(waves * 32, T), T, 0, stream>>>(
        (const bf16_t*)A, (const bf16_t*)Bt, C, M, N);
  };
  auto cvt = [&](const float* in, uint16_t* out, long long n) {
    k_f32_to_bf16<<<blk(n, T), T, 0, stream>>>(in, out, (int)n);
  };

  // ---- copy input features to layer-0 buffers + d_out columns [0:128) ----
  k_combine_store<<<blk((long long)NI * H, T), T, 0, stream>>>(
      x_item, nullptr, nullptr, hiA, out_items, NI, 0, 0);
  k_combine_store<<<blk((long long)NS * H, T), T, 0, stream>>>(
      x_sess, nullptr, nullptr, hsA, out_sess, NS, 0, 0);

  // ---- weight conversions (once) ----
  cvt(gg_Wih, w_ih, (long long)NL * H3 * H);     // used as A @ Wih^T -> Bt = Wih rows
  cvt(gg_Whh, w_hh, (long long)NL * H3 * H);
  for (int l = 0; l < NL; ++l) {
    k_transpose_bf16<<<blk(H * H, T), T, 0, stream>>>(gg_W + (size_t)l * H * H,
                                                      w_ggT + (size_t)l * H * H, H, H);
    k_transpose_bf16<<<blk(H * H, T), T, 0, stream>>>(gis_Wsrc + (size_t)l * H * H,
                                                      w_gis_s + (size_t)l * H * H, H, H);
    k_transpose_bf16<<<blk(H * H, T), T, 0, stream>>>(gis_Wdst + (size_t)l * H * H,
                                                      w_gis_d + (size_t)l * H * H, H, H);
    k_transpose_bf16<<<blk(H * H, T), T, 0, stream>>>(gsi_Wsrc + (size_t)l * H * H,
                                                      w_gsi_s + (size_t)l * H * H, H, H);
    k_transpose_bf16<<<blk(H * H, T), T, 0, stream>>>(gsi_Wdst + (size_t)l * H * H,
                                                      w_gsi_d + (size_t)l * H * H, H, H);
  }

  float* hi = hiA; float* hi_next = hiB;
  float* hs = hsA; float* hs_next = hsB;

  for (int l = 0; l < NL; ++l) {
    // bf16 staging of current node features
    cvt(hi, hi_bf, (long long)NI * H);
    cvt(hs, hs_bf, (long long)NS * H);

    // ---------- GGNN (item -> item) ----------
    gemm(hi_bf, w_ggT + (size_t)l * H * H, xw, NI, H);             // x @ W
    hipMemsetAsync(mseg, 0, (size_t)NI * H * 4, stream);
    k_seg_scatter<<<blk((long long)NE * 32, T), T, 0, stream>>>(xw, ii_src, ii_dst,
                                                                mseg, NE);
    cvt(mseg, m_bf, (long long)NI * H);
    gemm(m_bf, w_ih + (size_t)l * H3 * H, gi, NI, H3);             // m @ Wih^T
    gemm(hi_bf, w_hh + (size_t)l * H3 * H, gh, NI, H3);            // x @ Whh^T
    k_gru<<<blk((long long)NI * H, T), T, 0, stream>>>(
        gi, gh, gg_bih + (size_t)l * H3, gg_bhh + (size_t)l * H3, hi, xw,
        NI * H);                                                   // xw := GGNN out

    // ---------- GAT session -> item ----------
    gemm(hs_bf, w_gsi_s + (size_t)l * H * H, projS, NS, H);        // hs @ Wsrc
    gemm(hi_bf, w_gsi_d + (size_t)l * H * H, projD, NI, H);        // hi @ Wdst
    k_rowdot<<<blk((long long)NS * 32, T), T, 0, stream>>>(projS,
        gsi_asrc + (size_t)l * H, ssrc, NS);
    k_rowdot<<<blk((long long)NI * 32, T), T, 0, stream>>>(projD,
        gsi_adst + (size_t)l * H, sdst, NI);
    k_init_max_den<<<blk(NI, T), T, 0, stream>>>(mmax, den, NI);
    k_edge_logit_max<<<blk(NE, T), T, 0, stream>>>(ssrc, sdst, si_src, si_dst,
                                                   ebuf, mmax, NE);
    k_edge_exp_den<<<blk(NE, T), T, 0, stream>>>(ebuf, mmax, si_dst, den, NE);
    hipMemsetAsync(accI, 0, (size_t)NI * H * 4, stream);
    k_gat_scatter<<<blk((long long)NE * 32, T), T, 0, stream>>>(
        projS, ebuf, den, si_src, si_dst, accI, NE);

    // new item features = relu(GGNN + GAT_si + b)
    k_combine_store<<<blk((long long)NI * H, T), T, 0, stream>>>(
        xw, accI, gsi_b + (size_t)l * H, hi_next, out_items, NI, (l + 1) * H, 1);

    // ---------- GAT item -> session (uses pre-update hi, hs) ----------
    gemm(hi_bf, w_gis_s + (size_t)l * H * H, projS, NI, H);        // hi @ Wsrc
    gemm(hs_bf, w_gis_d + (size_t)l * H * H, projD, NS, H);        // hs @ Wdst
    k_rowdot<<<blk((long long)NI * 32, T), T, 0, stream>>>(projS,
        gis_asrc + (size_t)l * H, ssrc, NI);
    k_rowdot<<<blk((long long)NS * 32, T), T, 0, stream>>>(projD,
        gis_adst + (size_t)l * H, sdst, NS);
    k_init_max_den<<<blk(NS, T), T, 0, stream>>>(mmax, den, NS);
    k_edge_logit_max<<<blk(NE, T), T, 0, stream>>>(ssrc, sdst, is_src, is_dst,
                                                   ebuf, mmax, NE);
    k_edge_exp_den<<<blk(NE, T), T, 0, stream>>>(ebuf, mmax, is_dst, den, NE);
    hipMemsetAsync(accS, 0, (size_t)NS * H * 4, stream);
    k_gat_scatter<<<blk((long long)NE * 32, T), T, 0, stream>>>(
        projS, ebuf, den, is_src, is_dst, accS, NE);

    // new session features = relu(GAT_is + b)
    k_combine_store<<<blk((long long)NS * H, T), T, 0, stream>>>(
        accS, nullptr, gis_b + (size_t)l * H, hs_next, out_sess, NS, (l + 1) * H, 1);

    // ping-pong
    float* t;
    t = hi; hi = hi_next; hi_next = t;
    t = hs; hs = hs_next; hs_next = t;
  }
}
